// ParallelRealNVPNode_31121333027034
// MI455X (gfx1250) — compile-verified, matrix operand order checked by reading, not run on hardware
//
#include <hip/hip_runtime.h>
#include <hip/hip_bf16.h>

typedef __attribute__((ext_vector_type(16))) _Float16 v16h;
typedef __attribute__((ext_vector_type(8)))  _Float16 v8h;
typedef __attribute__((ext_vector_type(8)))  float    v8f;

#define B_SZ   4096
#define P_SZ   64
#define D_SZ   128
#define H_SZ   512
#define NS     64
#define NU     64
#define M_TILE 64
#define NTILES 32              // H_SZ / 16

#define W0_ELEMS (P_SZ * NS * H_SZ)     // 2,097,152 halves
#define W1_ELEMS (P_SZ * H_SZ * H_SZ)   // 16,777,216 halves
#define N_WG     ((B_SZ / M_TILE) * P_SZ)  // 4096 workgroups

// ---------------------------------------------------------------------------
// Weight pre-pack: fp32 (P,K,512) row-major -> f16 in WMMA B-operand order.
// Packed stream per (ks,nt) block: 32 lanes x 16 halves, where lane/half map
// to (K,N) per the CDNA5 16-bit operand layout:
//   k = (i>=8)*16 + (lane>=16)*8 + (i&7),  n = nt*16 + (lane&15)
// ---------------------------------------------------------------------------
__global__ __launch_bounds__(256) void pack_w_kernel(const float* __restrict__ src,
                                                     _Float16* __restrict__ dst,
                                                     int perP) {
  int tid = blockIdx.x * blockDim.x + threadIdx.x;
  int p = tid / perP;
  int r = tid - p * perP;
  int blk    = r >> 9;        // (ks*32 + nt)
  int within = r & 511;
  int lane   = within >> 4;
  int i      = within & 15;
  int ks = blk >> 5;
  int nt = blk & 31;
  int kk = ((i >= 8) ? 16 : 0) + ((lane >= 16) ? 8 : 0) + (i & 7);
  int k = ks * 32 + kk;
  int n = nt * 16 + (lane & 15);
  dst[tid] = (_Float16)src[p * perP + k * H_SZ + n];
}

// ---------------------------------------------------------------------------
// One WMMA layer: A (64 x 32*KS, packed f16 in LDS) @ B (32*KS x 512, packed
// f16 in global) + bias -> ReLU -> H (64 x 512, packed f16 in LDS).
// Wave w: m-tile = w&3, n-group = w>>2 covers 16 of the 32 n-tiles.
// ---------------------------------------------------------------------------
template <int KS>
__device__ __forceinline__ void gemm_layer(const _Float16* __restrict__ Alds,
                                           const _Float16* __restrict__ Bw,
                                           const float* __restrict__ bias,
                                           _Float16* __restrict__ Hout,
                                           int lane, int mt, int ng) {
  const int nn  = lane & 15;
  const int mhi = (lane >> 4) << 3;
  for (int j = 0; j < 16; ++j) {
    const int nt = ng + (j << 1);
    const float bv = bias[nt * 16 + nn];
    v8f acc = {bv, bv, bv, bv, bv, bv, bv, bv};
#pragma unroll
    for (int ks = 0; ks < KS; ++ks) {
      v16h a = *(const v16h*)(Alds + (mt * KS + ks) * 512 + lane * 16);
      v16h b = *(const v16h*)(Bw + (ks * NTILES + nt) * 512 + lane * 16);
      acc = __builtin_amdgcn_wmma_f32_16x16x32_f16(false, a, false, b,
                                                   (short)0, acc, false, false);
    }
    // ReLU + store into packed-A layout so H feeds the next layer directly.
    const int n0 = nt * 16 + nn;
    const int ks_o = n0 >> 5;
    const int kk = n0 & 31;
    const int laneA_base = ((kk >> 3) & 1) * 16;
    const int iA = (kk >> 4) * 8 + (kk & 7);
#pragma unroll
    for (int r = 0; r < 8; ++r) {
      const float v = fmaxf(acc[r], 0.0f);
      const int m = (mt << 4) + mhi + r;
      Hout[(mt * 16 + ks_o) * 512 + (laneA_base + (m & 15)) * 16 + iA] = (_Float16)v;
    }
  }
}

// ---------------------------------------------------------------------------
// Main fused kernel: one workgroup = (par p, 64-row batch tile).
// ---------------------------------------------------------------------------
__global__ __launch_bounds__(256) void realnvp_kernel(
    const float* __restrict__ x,
    const float* __restrict__ s_b0, const float* __restrict__ s_b1,
    const float* __restrict__ s_k2, const float* __restrict__ s_b2,
    const float* __restrict__ t_b0, const float* __restrict__ t_b1,
    const float* __restrict__ t_k2, const float* __restrict__ t_b2,
    const _Float16* __restrict__ sW0, const _Float16* __restrict__ sW1,
    const _Float16* __restrict__ tW0, const _Float16* __restrict__ tW1,
    float* __restrict__ out, float* __restrict__ partials) {
  extern __shared__ __align__(128) _Float16 smem[];
  _Float16* Xs = smem;                  // 64 x 64  f16 (packed A)   = 4096
  _Float16* H0 = smem + 4096;           // 64 x 512 f16 (packed A)   = 32768
  _Float16* H1 = H0 + 32768;            // 64 x 512 f16 (packed A)   = 32768
  float* svals = (float*)(H1 + 32768);  // 64
  float* tvals = svals + 64;            // 64
  float* esv   = tvals + 64;            // 64

  const int tid  = threadIdx.x;
  const int lane = tid & 31;
  const int wave = tid >> 5;
  const int mt   = wave & 3;
  const int ng   = wave >> 2;
  const int p    = blockIdx.y;
  const int b0   = blockIdx.x * M_TILE;

  // ---- Load x_static tile (dims 0..63) -> f16 packed-A layout in LDS.
  for (int e = tid; e < M_TILE * NS; e += 256) {
    const int m = e >> 6;
    const int k = e & 63;
    const float v = x[((long)(b0 + m) * P_SZ + p) * D_SZ + k];
    const int mt2 = m >> 4, ks = k >> 5, kk = k & 31;
    const int laneA = ((kk >> 3) & 1) * 16 + (m & 15);
    const int iA = (kk >> 4) * 8 + (kk & 7);
    Xs[(mt2 * 2 + ks) * 512 + laneA * 16 + iA] = (_Float16)v;
  }
  __syncthreads();

  // ---- Two MLPs (scale s, translate t) sharing LDS scratch.
  for (int which = 0; which < 2; ++which) {
    const _Float16* W0 = (which ? tW0 : sW0) + (long)p * (NS * H_SZ);
    const _Float16* W1 = (which ? tW1 : sW1) + (long)p * (H_SZ * H_SZ);
    const float* bb0 = (which ? t_b0 : s_b0) + p * H_SZ;
    const float* bb1 = (which ? t_b1 : s_b1) + p * H_SZ;
    const float* w2  = (which ? t_k2 : s_k2) + p * H_SZ;
    const float  b2v = (which ? t_b2 : s_b2)[p];
    float* outv = which ? tvals : svals;

    gemm_layer<2>(Xs, W0, bb0, H0, lane, mt, ng);   // layer 0: K=64
    __syncthreads();
    gemm_layer<16>(H0, W1, bb1, H1, lane, mt, ng);  // layer 1: K=512
    __syncthreads();

    // layer 2: per-row dot(H1[row,:], w2) + b2 (N=1 -> vectorized reduction).
    // In the packed-A layout, 8 consecutive n within an aligned group share
    // laneA and have contiguous iA -> read H1 as 16-byte v8h vectors.
    {
      const int row = tid >> 2;
      const int q = tid & 3;
      float sum = 0.f;
      const int mt2 = row >> 4;
      const int rlo = row & 15;
      for (int g = q * 16; g < q * 16 + 16; ++g) {
        const int n0 = g << 3;
        const int ks = n0 >> 5;
        const int kk0 = n0 & 31;
        const int laneA = ((kk0 >> 3) & 1) * 16 + rlo;
        const v8h hv = *(const v8h*)(H1 + (mt2 * 16 + ks) * 512 + laneA * 16 +
                                     ((kk0 >> 4) << 3));
        const float4 w2a = *(const float4*)(w2 + n0);
        const float4 w2b = *(const float4*)(w2 + n0 + 4);
        sum += (float)hv[0] * w2a.x + (float)hv[1] * w2a.y +
               (float)hv[2] * w2a.z + (float)hv[3] * w2a.w;
        sum += (float)hv[4] * w2b.x + (float)hv[5] * w2b.y +
               (float)hv[6] * w2b.z + (float)hv[7] * w2b.w;
      }
      sum += __shfl_xor(sum, 1, 32);
      sum += __shfl_xor(sum, 2, 32);
      if (q == 0) outv[row] = sum + b2v;
    }
    __syncthreads();
  }

  // ---- Per-WG partial sum of s (deterministic reduction in pass 2).
  if (tid == 0) {
    float acc = 0.f;
    for (int m = 0; m < M_TILE; ++m) acc += svals[m];
    partials[blockIdx.y * gridDim.x + blockIdx.x] = acc;
  }
  if (tid < M_TILE) esv[tid] = __expf(svals[tid]);
  __syncthreads();

  // ---- Epilogue: out[:, :64] = x static; out[:, 64:] = x_upd*exp(s) + t.
  for (int e = tid; e < M_TILE * D_SZ; e += 256) {
    const int m = e >> 7;
    const int d = e & 127;
    const long gi = ((long)(b0 + m) * P_SZ + p) * D_SZ + d;
    const float xv = x[gi];
    out[gi] = (d < NS) ? xv : (xv * esv[m] + tvals[m]);
  }
}

// ---------------------------------------------------------------------------
// Deterministic final reduction of log_det_jac.
// ---------------------------------------------------------------------------
__global__ __launch_bounds__(256) void reduce_ldj_kernel(const float* __restrict__ partials,
                                                         const float* __restrict__ ldj_in,
                                                         float* __restrict__ ldj_out) {
  __shared__ float buf[256];
  float s = 0.f;
  for (int i = threadIdx.x; i < N_WG; i += 256) s += partials[i];
  buf[threadIdx.x] = s;
  __syncthreads();
  for (int st = 128; st > 0; st >>= 1) {
    if (threadIdx.x < st) buf[threadIdx.x] += buf[threadIdx.x + st];
    __syncthreads();
  }
  if (threadIdx.x == 0) *ldj_out = buf[0] + ldj_in[0];
}

// ---------------------------------------------------------------------------
extern "C" void kernel_launch(void* const* d_in, const int* in_sizes, int n_in,
                              void* d_out, int out_size, void* d_ws, size_t ws_size,
                              hipStream_t stream) {
  const float* s_k0 = (const float*)d_in[0];
  const float* s_b0 = (const float*)d_in[1];
  const float* s_k1 = (const float*)d_in[2];
  const float* s_b1 = (const float*)d_in[3];
  const float* s_k2 = (const float*)d_in[4];
  const float* s_b2 = (const float*)d_in[5];
  const float* t_k0 = (const float*)d_in[6];
  const float* t_b0 = (const float*)d_in[7];
  const float* t_k1 = (const float*)d_in[8];
  const float* t_b1 = (const float*)d_in[9];
  const float* t_k2 = (const float*)d_in[10];
  const float* t_b2 = (const float*)d_in[11];
  const float* x    = (const float*)d_in[12];
  const float* ldj  = (const float*)d_in[13];

  _Float16* sW0 = (_Float16*)d_ws;
  _Float16* sW1 = sW0 + W0_ELEMS;
  _Float16* tW0 = sW1 + W1_ELEMS;
  _Float16* tW1 = tW0 + W0_ELEMS;
  float* partials = (float*)(tW1 + W1_ELEMS);  // N_WG floats

  float* out = (float*)d_out;
  float* ldj_out = out + (size_t)B_SZ * P_SZ * D_SZ;

  pack_w_kernel<<<W0_ELEMS / 256, 256, 0, stream>>>(s_k0, sW0, NS * H_SZ);
  pack_w_kernel<<<W1_ELEMS / 256, 256, 0, stream>>>(s_k1, sW1, H_SZ * H_SZ);
  pack_w_kernel<<<W0_ELEMS / 256, 256, 0, stream>>>(t_k0, tW0, NS * H_SZ);
  pack_w_kernel<<<W1_ELEMS / 256, 256, 0, stream>>>(t_k1, tW1, H_SZ * H_SZ);

  const size_t shmem = (4096 + 32768 + 32768) * sizeof(_Float16) + 3 * 64 * sizeof(float);
  dim3 grid(B_SZ / M_TILE, P_SZ);
  realnvp_kernel<<<grid, 256, shmem, stream>>>(x, s_b0, s_b1, s_k2, s_b2,
                                               t_b0, t_b1, t_k2, t_b2,
                                               sW0, sW1, tW0, tW1, out, partials);

  reduce_ldj_kernel<<<1, 256, 0, stream>>>(partials, ldj, ldj_out);
}
